// HyperAttnBlock_78297253806430
// MI455X (gfx1250) — matrix operand order repaired and checked by reading, not compiled, not run on hardware
//
#include <hip/hip_runtime.h>

typedef __attribute__((ext_vector_type(16))) __bf16 v16bf;
typedef __attribute__((ext_vector_type(8)))  float  v8f;

constexpr int kC = 192, kT = 128, kV = 25, kH = 6, kDIM = 192;
constexpr float kScale = 0.17677669529663687f; // 1/sqrt(32)

__device__ inline unsigned short f2bf(float f) {
  unsigned u = __float_as_uint(f);
  u += 0x7FFFu + ((u >> 16) & 1u);   // round-to-nearest-even
  return (unsigned short)(u >> 16);
}
__device__ inline float bf2f(unsigned short h) {
  return __uint_as_float(((unsigned)h) << 16);
}

// Packed fp32x2 -> bf16x2: prefer HW packed cvt; else 2x v_add + v_perm_b32
// (round-to-nearest, ties away) -- 3 VALU ops per pair instead of ~10.
__device__ inline unsigned cvt2bf(float lo, float hi) {
#if __has_builtin(__builtin_amdgcn_cvt_pk_bf16_f32)
  auto p = __builtin_amdgcn_cvt_pk_bf16_f32(lo, hi);
  unsigned u;
  __builtin_memcpy(&u, &p, 4);
  return u;
#elif __has_builtin(__builtin_amdgcn_perm)
  unsigned a = __float_as_uint(lo) + 0x8000u;
  unsigned b = __float_as_uint(hi) + 0x8000u;
  // result bytes (LSB..MSB) = {a.b2, a.b3, b.b2, b.b3}; src1=a -> idx 0..3, src0=b -> idx 4..7
  return __builtin_amdgcn_perm(b, a, 0x07060302u);
#else
  return (unsigned)f2bf(lo) | ((unsigned)f2bf(hi) << 16);
#endif
}

union Frag { v16bf v; unsigned u[8]; };

// A-matrix fragment (16x32 bf16, M x K), row-major source with K contiguous.
// Lane L: M = r0 + L%16 ; VGPR i holds K pair at ((i<4)?0:16) + 8*(L/16) + 2*(i&3).
__device__ inline v16bf load_fragA(const unsigned short* p, int stride, int r0, int k0, int lane) {
  Frag f;
  const unsigned short* base = p + (r0 + (lane & 15)) * stride + k0;
  int g = (lane >> 4) & 1;
#pragma unroll
  for (int i = 0; i < 8; ++i) {
    int k = ((i < 4) ? 0 : 16) + g * 8 + (i & 3) * 2;
    f.u[i] = *(const unsigned*)(base + k);
  }
  return f.v;
}

// B-matrix fragment (32x16 bf16, K x N), source stored as B^T row-major (K contiguous per n).
// Lane L: N = n0 + L%16 ; VGPR i holds K pair at ((i<4)?0:16) + 8*(L/16) + 2*(i&3).
__device__ inline v16bf load_fragB(const unsigned short* p, int stride, int n0, int k0, int lane) {
  Frag f;
  const unsigned short* base = p + (n0 + (lane & 15)) * stride + k0;
  int g = (lane >> 4) & 1;
#pragma unroll
  for (int i = 0; i < 8; ++i) {
    int k = ((i < 4) ? 0 : 16) + g * 8 + (i & 3) * 2;
    f.u[i] = *(const unsigned*)(base + k);
  }
  return f.v;
}

__device__ inline v8f wmma_bf16(v16bf a, v16bf b, v8f c) {
  return __builtin_amdgcn_wmma_f32_16x16x32_bf16(false, a, false, b, (short)0, c, false, false);
}

// ---- weight prep: fp32 -> bf16 once into workspace (stays hot in 192MB L2) ----
__global__ void hyperattn_prep(const float* __restrict__ qw, const float* __restrict__ kvw,
                               const float* __restrict__ pw,
                               unsigned short* __restrict__ qwb,
                               unsigned short* __restrict__ vwb,
                               unsigned short* __restrict__ pwb) {
  int i = (blockIdx.x * 256 + threadIdx.x) * 2;  // element pair
  if (i < kDIM * kC) {
    *(unsigned*)&qwb[i] = cvt2bf(qw[i], qw[i + 1]);
    *(unsigned*)&vwb[i] = cvt2bf(kvw[kDIM * kC + i], kvw[kDIM * kC + i + 1]);
    *(unsigned*)&pwb[i] = cvt2bf(pw[i], pw[i + 1]);
  }
}

// ---- fused block: one workgroup per (n,t) ----
__global__ __launch_bounds__(256)
void hyperattn_main(const float* __restrict__ x, const float* __restrict__ e,
                    const float* __restrict__ w1, const float* __restrict__ alpha,
                    const float* __restrict__ outer, const float* __restrict__ proj_b,
                    const unsigned short* __restrict__ qwb,
                    const unsigned short* __restrict__ vwb,
                    const unsigned short* __restrict__ pwb,
                    float* __restrict__ out) {
  // LDS aliasing across phases keeps us ~61 KB:
  //   xs_t [P0..P1] and out_t [P4..P5] overlap ct [P2..P3]
  //   q_t  [P1..P2] overlaps ms [P3..P4]
  union U1 {
    struct { unsigned short xs_t[32 * 192]; unsigned short out_t[32 * 192]; } s;
    float ct[kH * 32 * 32];
  };
  union U2 {
    unsigned short q_t[32 * 192];
    unsigned short ms[kH * 32 * 32];
  };
  __shared__ __align__(16) U1 u1;
  __shared__ __align__(16) U2 u2;
  __shared__ __align__(16) unsigned short es_t[32 * 192]; // e slice, [v][c] (c contiguous)
  __shared__ __align__(16) unsigned short vs[192 * 32];   // vv,      [o][vk] (vk contiguous)
  __shared__ float dt[kH * 32];                           // dterm per head

  const int tid = threadIdx.x, lane = tid & 31, wave = tid >> 5;
  const int blk = blockIdx.x, n = blk / kT, t = blk % kT;
  const size_t base = (size_t)n * kC * kT * kV + (size_t)t * kV;
  const int cstr = kT * kV; // 3200

  // P0: load x/e slices -> bf16 (packed cvt on v-pairs), transposed, zero-padded cols 25..31
  for (int i = tid; i < 192 * 16; i += 256) {
    int c = i >> 4, vp = i & 15;
    int v0 = vp * 2, v1 = v0 + 1;
    size_t off = base + (size_t)c * cstr + (size_t)v0;
    float x0 = (v0 < kV) ? x[off] : 0.f;
    float x1 = (v1 < kV) ? x[off + 1] : 0.f;
    float e0 = (v0 < kV) ? e[off] : 0.f;
    float e1 = (v1 < kV) ? e[off + 1] : 0.f;
    unsigned px = cvt2bf(x0, x1);
    unsigned pe = cvt2bf(e0, e1);
    u1.s.xs_t[v0 * 192 + c] = (unsigned short)px;
    u1.s.xs_t[v1 * 192 + c] = (unsigned short)(px >> 16);
    es_t[v0 * 192 + c] = (unsigned short)pe;
    es_t[v1 * 192 + c] = (unsigned short)(pe >> 16);
  }
  __syncthreads();

  // dterm[h][vk] = sum_hd w1[h,hd] * e_k[vk,hd]   (cheap, overlaps P1)
  if (tid < 192) {
    int h = tid >> 5, vk = tid & 31;
    float s = 0.f;
    if (vk < kV) {
#pragma unroll
      for (int hd = 0; hd < 32; ++hd)
        s += w1[h * 32 + hd] * bf2f(es_t[vk * 192 + h * 32 + hd]);
    }
    dt[tid] = s;
  }

  // P1: q = q_w @ x  and  vv = kv_w[v] @ x   (192x32 out, K=192)
  for (int j = wave; j < 48; j += 8) {
    const unsigned short* W = (j < 24) ? qwb : vwb;
    int tile = (j < 24) ? j : j - 24;
    int m0 = (tile >> 1) * 16, n0 = (tile & 1) * 16;
    v8f acc = {};
#pragma unroll
    for (int k0 = 0; k0 < 192; k0 += 32) {
      v16bf a = load_fragA(W, 192, m0, k0, lane);            // weights via L2
      v16bf b = load_fragB(u1.s.xs_t, 192, n0, k0, lane);    // LDS
      acc = wmma_bf16(a, b, acc);
    }
    int nn = n0 + (lane & 15), mb = m0 + 8 * (lane >> 4);
    if (j < 24) { // q -> q_t[v][o] (o contiguous): next stage uses q as A over K=hd
#pragma unroll
      for (int r = 0; r < 8; r += 2)
        *(unsigned*)&u2.q_t[nn * 192 + mb + r] = cvt2bf(acc[r], acc[r + 1]);
    } else {      // vv -> vs[o][vk] (vk contiguous): used as A over K=vk
#pragma unroll
      for (int r = 0; r < 8; r += 2) {
        unsigned p = cvt2bf(acc[r], acc[r + 1]);
        vs[(mb + r) * 32 + nn] = (unsigned short)p;
        vs[(mb + r + 1) * 32 + nn] = (unsigned short)(p >> 16);
      }
    }
  }
  __syncthreads();

  // P2: cterm[h][vq][vk] = q_h @ e_h^T  (K = hd = 32, single WMMA per 16x16 tile)
  for (int j = wave; j < 24; j += 8) {
    int h = j >> 2, m0 = ((j >> 1) & 1) * 16, n0 = (j & 1) * 16;
    v16bf a = load_fragA(u2.q_t, 192, m0, h * 32, lane);
    v16bf b = load_fragB(es_t, 192, n0, h * 32, lane);
    v8f acc = {};
    acc = wmma_bf16(a, b, acc);
    int nn = n0 + (lane & 15), mb = m0 + 8 * (lane >> 4);
    float* ctb = &u1.ct[h * 1024];
#pragma unroll
    for (int r = 0; r < 8; ++r) ctb[(mb + r) * 32 + nn] = acc[r];
  }
  __syncthreads();

  // P3: softmax over vk (25), then m = alpha*attn + outer  -> bf16 ms[h][vq][vk]
  if (tid < 192) {
    int h = tid >> 5, vq = tid & 31;
    unsigned* mrow = (unsigned*)&u2.ms[(h * 32 + vq) * 32];
    if (vq < kV) {
      const float* ctb = &u1.ct[(h * 32 + vq) * 32];
      const float* dh = &dt[h * 32];
      float row[kV];
      float mx = -3.0e38f;
#pragma unroll
      for (int k = 0; k < kV; ++k) {
        float z = (ctb[k] + dh[k]) * kScale;
        row[k] = z;
        mx = fmaxf(mx, z);
      }
      float s = 0.f;
#pragma unroll
      for (int k = 0; k < kV; ++k) {
        float p = __expf(row[k] - mx);
        row[k] = p;
        s += p;
      }
      float ia = alpha[0] / s;
      const float* orow = &outer[(h * kV + vq) * kV];
#pragma unroll
      for (int kp = 0; kp < 12; ++kp)  // vk 0..23 as pairs
        mrow[kp] = cvt2bf(row[2 * kp] * ia + orow[2 * kp],
                          row[2 * kp + 1] * ia + orow[2 * kp + 1]);
      mrow[12] = cvt2bf(row[24] * ia + orow[24], 0.f);  // vk 24 + pad
#pragma unroll
      for (int kp = 13; kp < 16; ++kp) mrow[kp] = 0u;   // vk 26..31 pad
    } else {
#pragma unroll
      for (int kp = 0; kp < 16; ++kp) mrow[kp] = 0u;
    }
  }
  __syncthreads();

  // P4: out_block[hd][vq] = vv_h @ m_h^T  (K = vk padded to 32) -> out_t[vq][c]
  for (int j = wave; j < 24; j += 8) {
    int h = j >> 2, m0 = ((j >> 1) & 1) * 16, n0 = (j & 1) * 16;
    v16bf a = load_fragA(&vs[h * 32 * 32], 32, m0, 0, lane);
    v16bf b = load_fragB(&u2.ms[h * 1024], 32, n0, 0, lane);
    v8f acc = {};
    acc = wmma_bf16(a, b, acc);
    int nn = n0 + (lane & 15), mb = h * 32 + m0 + 8 * (lane >> 4);
#pragma unroll
    for (int r = 0; r < 8; r += 2)
      *(unsigned*)&u1.s.out_t[nn * 192 + mb + r] = cvt2bf(acc[r], acc[r + 1]);
  }
  __syncthreads();

  // P5: final = proj_w @ out_mat + proj_b  (192x32, K=192), strided store to [N,DIM,T,V]
  for (int j = wave; j < 24; j += 8) {
    int m0 = (j >> 1) * 16, n0 = (j & 1) * 16;
    v8f acc = {};
#pragma unroll
    for (int k0 = 0; k0 < 192; k0 += 32) {
      v16bf a = load_fragA(pwb, 192, m0, k0, lane);
      v16bf b = load_fragB(u1.s.out_t, 192, n0, k0, lane);
      acc = wmma_bf16(a, b, acc);
    }
    int nn = n0 + (lane & 15), mb = m0 + 8 * (lane >> 4);
    if (nn < kV) {
#pragma unroll
      for (int r = 0; r < 8; ++r) {
        size_t off = (size_t)n * kDIM * kT * kV + (size_t)(mb + r) * kT * kV + (size_t)t * kV + nn;
        out[off] = acc[r] + proj_b[mb + r];
      }
    }
  }
}

extern "C" void kernel_launch(void* const* d_in, const int* in_sizes, int n_in,
                              void* d_out, int out_size, void* d_ws, size_t ws_size,
                              hipStream_t stream) {
  const float* x      = (const float*)d_in[0];
  const float* e      = (const float*)d_in[1];
  const float* q_w    = (const float*)d_in[2];
  const float* kv_w   = (const float*)d_in[3];
  const float* w1     = (const float*)d_in[4];
  const float* alpha  = (const float*)d_in[5];
  const float* outer  = (const float*)d_in[6];
  const float* proj_w = (const float*)d_in[7];
  const float* proj_b = (const float*)d_in[8];
  float* out = (float*)d_out;

  unsigned short* ws  = (unsigned short*)d_ws;
  unsigned short* qwb = ws;
  unsigned short* vwb = ws + kDIM * kC;
  unsigned short* pwb = ws + 2 * kDIM * kC;

  hipLaunchKernelGGL(hyperattn_prep, dim3((kDIM * kC / 2 + 255) / 256), dim3(256), 0, stream,
                     q_w, kv_w, proj_w, qwb, vwb, pwb);
  hipLaunchKernelGGL(hyperattn_main, dim3(64 * kT), dim3(256), 0, stream,
                     x, e, w1, alpha, outer, proj_b, qwb, vwb, pwb, out);
}
